// TransformerBlock_47047071760675
// MI455X (gfx1250) — matrix operand order repaired and checked by reading, not compiled, not run on hardware
//
#include <hip/hip_runtime.h>
#include <hip/hip_bf16.h>

// ---------------------------------------------------------------------------
// Transformer block forward (pre-LN, BERT-style) for MI455X (gfx1250).
// B=2, S=2048, D=768, H=12, HD=64, FFN=3072. All GEMMs via
// v_wmma_f32_16x16x32_bf16 (bf16 in, fp32 accumulate). Flash-attention style
// online softmax so the S x S score tensor is never materialized.
// GEMM waves compute 32x64 output tiles (8 WMMAs per 32-wide K step) and
// prefetch the A stream with global_prefetch_b8.
// ---------------------------------------------------------------------------

typedef __attribute__((ext_vector_type(16))) __bf16 v16bf;
typedef __attribute__((ext_vector_type(8)))  __bf16 v8bf;
typedef __attribute__((ext_vector_type(8)))  float  v8f;

#define D_   768
#define H_   12
#define B_   2
#define S_   2048
#define HD_  64
#define DFF_ 3072
#define ROWS_ (B_ * S_)          // 4096

static __device__ __forceinline__ v8f wmma_bf16(v16bf a, v16bf b, v8f c) {
  return __builtin_amdgcn_wmma_f32_16x16x32_bf16(
      /*neg_a=*/false, a, /*neg_b=*/false, b,
      /*c_mod=*/(short)0, c, /*reuse_a=*/false, /*reuse_b=*/false);
}

static __device__ __forceinline__ v16bf load_afrag(const __bf16* row_ptr,
                                                   int koff) {
  v8bf a0 = *(const v8bf*)(row_ptr + koff);
  v8bf a1 = *(const v8bf*)(row_ptr + 8 + koff);
  v16bf a;
#pragma unroll
  for (int i = 0; i < 8; ++i) { a[i] = a0[i]; a[8 + i] = a1[i]; }
  return a;
}

// ---------------------------------------------------------------------------
// LayerNorm: one wave per row of length D_=768 (24 elems / lane), bf16 out.
// ---------------------------------------------------------------------------
__global__ void __launch_bounds__(256)
k_layernorm(const float* __restrict__ x, const float* __restrict__ scale,
            const float* __restrict__ shift, __bf16* __restrict__ out) {
  int wave = threadIdx.x >> 5, lane = threadIdx.x & 31;
  int row  = blockIdx.x * 8 + wave;
  const float* xr = x + (size_t)row * D_;
  float vals[24];
  float s = 0.f;
#pragma unroll
  for (int i = 0; i < 24; ++i) { vals[i] = xr[lane + 32 * i]; s += vals[i]; }
#pragma unroll
  for (int m = 16; m >= 1; m >>= 1) s += __shfl_xor(s, m, 32);
  float mean = s * (1.0f / D_);
  float v = 0.f;
#pragma unroll
  for (int i = 0; i < 24; ++i) { float d = vals[i] - mean; v += d * d; }
#pragma unroll
  for (int m = 16; m >= 1; m >>= 1) v += __shfl_xor(v, m, 32);
  float rstd = rsqrtf(v * (1.0f / D_) + 1e-5f);
  __bf16* outr = out + (size_t)row * D_;
#pragma unroll
  for (int i = 0; i < 24; ++i) {
    int c = lane + 32 * i;
    outr[c] = (__bf16)(scale[c] * ((vals[i] - mean) * rstd) + shift[c]);
  }
}

// ---------------------------------------------------------------------------
// Pack fp32 weight W (K x N, row major) into bf16 WMMA B-fragment tiles.
// Tile = (kt, nt) covering K 32 x N 16. Tile storage: [lane][8 vgpr pairs]
// (16 bf16 per lane, contiguous) so a wave fetches its fragment with one
// 32-byte load. Tile order: nt-major, kt-minor (streaming along K).
// ---------------------------------------------------------------------------
__global__ void __launch_bounds__(256)
k_pack(const float* __restrict__ W, __bf16* __restrict__ out,
       int N, int Ktiles, int total) {
  int t = blockIdx.x * 256 + threadIdx.x;
  if (t >= total) return;
  int e = t & 511, tile = t >> 9;
  int lane = e >> 4, idx = e & 15;
  int j = idx >> 1, p = idx & 1;
  int nt = tile / Ktiles, kt = tile - nt * Ktiles;
  int kk = kt * 32 + ((lane >= 16) ? 16 : 0) + 2 * j + p;
  int nn = nt * 16 + (lane & 15);
  out[t] = (__bf16)W[(size_t)kk * N + nn];
}

// ---------------------------------------------------------------------------
// GEMM: C(M x N) = A(M x K, bf16 row-major) * Bp(packed tiles) [+bias][gelu]
// [+resid]. 8 waves/block; each wave computes 32(M) x 64(N): per K-step it
// loads 2 A fragments + 4 B fragments and issues 8 WMMAs (21 FLOP/byte of
// fragment traffic). Block tile: 256 x 64. Exact fit, no guards -> EXEC
// stays all-ones around WMMA as the ISA requires.
// ---------------------------------------------------------------------------
__global__ void __launch_bounds__(256)
k_gemm(const __bf16* __restrict__ A, const __bf16* __restrict__ Bp,
       int K, int N, const float* __restrict__ bias,
       const float* __restrict__ resid, int gelu,
       __bf16* __restrict__ outB, float* __restrict__ outF) {
  int wave = threadIdx.x >> 5, lane = threadIdx.x & 31;
  int m0 = blockIdx.x * 256 + wave * 32;
  int n0 = blockIdx.y * 64;
  int row  = lane & 15;
  int koff = (lane >= 16) ? 16 : 0;
  int Ktiles = K >> 5;

  v8f acc[2][4];
#pragma unroll
  for (int u = 0; u < 2; ++u)
#pragma unroll
    for (int t = 0; t < 4; ++t)
#pragma unroll
      for (int i = 0; i < 8; ++i) acc[u][t][i] = 0.f;

  const __bf16* arow0 = A + (size_t)(m0 + row) * K;
  const __bf16* arow1 = A + (size_t)(m0 + 16 + row) * K;
  for (int kt = 0; kt < Ktiles; ++kt) {
    // pull the A stream ahead while WMMAs execute (global_prefetch_b8)
    __builtin_prefetch(arow0 + kt * 32 + 1024, 0, 1);
    __builtin_prefetch(arow1 + kt * 32 + 1024, 0, 1);
    v16bf a0 = load_afrag(arow0 + kt * 32, koff);
    v16bf a1 = load_afrag(arow1 + kt * 32, koff);
#pragma unroll
    for (int t = 0; t < 4; ++t) {
      const __bf16* bt =
          Bp + ((size_t)((n0 >> 4) + t) * Ktiles + kt) * 512 + lane * 16;
      v16bf b = *(const v16bf*)bt;
      acc[0][t] = wmma_bf16(a0, b, acc[0][t]);
      acc[1][t] = wmma_bf16(a1, b, acc[1][t]);
    }
  }

  int half = (lane >= 16) ? 8 : 0;
#pragma unroll
  for (int u = 0; u < 2; ++u) {
#pragma unroll
    for (int t = 0; t < 4; ++t) {
      int col = n0 + t * 16 + (lane & 15);
      float bv = bias ? bias[col] : 0.f;
#pragma unroll
      for (int r = 0; r < 8; ++r) {
        int mrow = m0 + u * 16 + r + half;
        float vv = acc[u][t][r] + bv;
        if (gelu) vv = 0.5f * vv * (1.0f + erff(vv * 0.70710678f));
        if (resid) vv += resid[(size_t)mrow * N + col];
        if (outF) outF[(size_t)mrow * N + col] = vv;
        else      outB[(size_t)mrow * N + col] = (__bf16)vv;
      }
    }
  }
}

// ---------------------------------------------------------------------------
// Transpose V (B,S,H,HD) bf16 -> Vt (B,H,HD,S) bf16 so the PV B-fragments
// become contiguous loads along S.
// ---------------------------------------------------------------------------
__global__ void __launch_bounds__(256)
k_transpose_v(const __bf16* __restrict__ v, __bf16* __restrict__ vt) {
  int t = blockIdx.x * 256 + threadIdx.x;     // total = B*H*HD*S
  int srow = t & (S_ - 1);
  int rest = t >> 11;                          // / S_
  int d  = rest & (HD_ - 1);
  int bh = rest >> 6;                          // b*H + h
  int b = bh / H_, h = bh - b * H_;
  vt[t] = v[(size_t)(b * S_ + srow) * D_ + h * HD_ + d];
}

// ---------------------------------------------------------------------------
// Flash attention: one wave per (b, h, 16-row q tile). KV loop in steps of
// 32: 4 WMMAs for scores (2 kv subtiles x 2 k-steps over HD=64), online
// softmax via half-wave shfl reductions, P transposed C->A layout through
// wave-private LDS, then 4 WMMAs for P*V against pre-transposed Vt.
// ---------------------------------------------------------------------------
__global__ void __launch_bounds__(256)
k_attn(const __bf16* __restrict__ qb, const __bf16* __restrict__ kb,
       const __bf16* __restrict__ vtb, const long long* __restrict__ qmask,
       __bf16* __restrict__ ctxb) {
  __shared__ __bf16 Pl[8 * 512];               // 8 waves x (16x32) bf16
  int wave = threadIdx.x >> 5, lane = threadIdx.x & 31;
  int task = blockIdx.x * 8 + wave;            // exactly B*H*(S/16) = 3072
  int qt = task & (S_ / 16 - 1);
  int bh = task >> 7;
  int b = bh / H_, h = bh - b * H_;
  int m0 = qt * 16;
  int row  = lane & 15;
  int koff = (lane >= 16) ? 16 : 0;
  int hi16 = (lane >= 16) ? 16 : 0;
  int half = (lane >= 16) ? 8 : 0;

  // Q fragments for the two 32-wide k-steps across HD=64.
  const __bf16* qrow = qb + (size_t)(b * S_ + m0 + row) * D_ + h * HD_;
  v16bf qf[2];
#pragma unroll
  for (int ks = 0; ks < 2; ++ks) qf[ks] = load_afrag(qrow + ks * 32, koff);

  float mr[8], lr[8];
  v8f acc[4];
#pragma unroll
  for (int r = 0; r < 8; ++r) { mr[r] = -3.0e38f; lr[r] = 0.f; }
#pragma unroll
  for (int t = 0; t < 4; ++t)
#pragma unroll
    for (int i = 0; i < 8; ++i) acc[t][i] = 0.f;

  __bf16* pw = Pl + wave * 512;
  const long long* mk = qmask + b * S_;

  for (int kv = 0; kv < S_; kv += 32) {
    v8f s0, s1;
#pragma unroll
    for (int i = 0; i < 8; ++i) { s0[i] = 0.f; s1[i] = 0.f; }
#pragma unroll
    for (int ks = 0; ks < 2; ++ks) {
      v16bf b0 = *(const v16bf*)(kb + (size_t)(b * S_ + kv + row) * D_ +
                                 h * HD_ + ks * 32 + hi16);
      v16bf b1 = *(const v16bf*)(kb + (size_t)(b * S_ + kv + 16 + row) * D_ +
                                 h * HD_ + ks * 32 + hi16);
      s0 = wmma_bf16(qf[ks], b0, s0);
      s1 = wmma_bf16(qf[ks], b1, s1);
    }
    bool km0 = mk[kv + row] != 0;        // key-column mask (lane = column)
    bool km1 = mk[kv + 16 + row] != 0;
#pragma unroll
    for (int r = 0; r < 8; ++r) {
      float f0 = km0 ? s0[r] * 0.125f : -3.0e38f;   // 1/sqrt(64)
      float f1 = km1 ? s1[r] * 0.125f : -3.0e38f;
      float mx = fmaxf(f0, f1);
#pragma unroll
      for (int d = 8; d >= 1; d >>= 1) mx = fmaxf(mx, __shfl_xor(mx, d, 32));
      float mn = fmaxf(mr[r], mx);
      float alpha = __expf(mr[r] - mn);
      mr[r] = mn;
      float e0 = __expf(f0 - mn), e1 = __expf(f1 - mn);
      float rs = e0 + e1;
#pragma unroll
      for (int d = 8; d >= 1; d >>= 1) rs += __shfl_xor(rs, d, 32);
      lr[r] = lr[r] * alpha + rs;
#pragma unroll
      for (int t = 0; t < 4; ++t) acc[t][r] *= alpha;
      // stage P (C-layout -> LDS row-major 16x32)
      pw[(r + half) * 32 + (lane & 15)]      = (__bf16)e0;
      pw[(r + half) * 32 + 16 + (lane & 15)] = (__bf16)e1;
    }
    // reload P as an A-fragment (wave-private; DS ops are in-order per wave)
    v16bf pf = load_afrag(pw + row * 32, koff);
#pragma unroll
    for (int t = 0; t < 4; ++t) {
      const __bf16* vtp =
          vtb + (size_t)(bh * HD_ + t * 16 + (lane & 15)) * S_ + kv + hi16;
      v16bf vf = *(const v16bf*)vtp;
      acc[t] = wmma_bf16(pf, vf, acc[t]);
    }
  }

#pragma unroll
  for (int r = 0; r < 8; ++r) {
    float inv = 1.0f / lr[r];
    __bf16* op = ctxb + (size_t)(b * S_ + m0 + r + half) * D_ + h * HD_;
#pragma unroll
    for (int t = 0; t < 4; ++t)
      op[t * 16 + (lane & 15)] = (__bf16)(acc[t][r] * inv);
  }
}

// ---------------------------------------------------------------------------
// Host-side orchestration.
// ---------------------------------------------------------------------------
extern "C" void kernel_launch(void* const* d_in, const int* in_sizes, int n_in,
                              void* d_out, int out_size, void* d_ws,
                              size_t ws_size, hipStream_t stream) {
  const float*     q     = (const float*)d_in[0];
  const long long* qmask = (const long long*)d_in[1];
  const float*     Wq    = (const float*)d_in[2];
  const float*     Wk    = (const float*)d_in[3];
  const float*     Wv    = (const float*)d_in[4];
  const float*     Wo    = (const float*)d_in[5];
  const float*     bo    = (const float*)d_in[6];
  const float*     ln1s  = (const float*)d_in[7];
  const float*     ln1b  = (const float*)d_in[8];
  const float*     ln2s  = (const float*)d_in[9];
  const float*     ln2b  = (const float*)d_in[10];
  const float*     W1    = (const float*)d_in[11];
  const float*     b1    = (const float*)d_in[12];
  const float*     W2    = (const float*)d_in[13];
  const float*     b2    = (const float*)d_in[14];
  (void)in_sizes; (void)n_in; (void)out_size; (void)ws_size;

  char* ws = (char*)d_ws;
  size_t off = 0;
  auto take = [&](size_t bytes) -> char* {
    char* p = ws + off;
    off += (bytes + 255) & ~(size_t)255;
    return p;
  };
  const size_t actB = (size_t)ROWS_ * D_ * sizeof(__bf16);    // 6.29 MB
  const size_t wDB  = (size_t)D_ * D_ * sizeof(__bf16);       // 1.18 MB
  const size_t wFB  = (size_t)D_ * DFF_ * sizeof(__bf16);     // 4.72 MB
  const size_t ffB  = (size_t)ROWS_ * DFF_ * sizeof(__bf16);  // 25.2 MB

  __bf16* xn   = (__bf16*)take(actB);
  __bf16* pWq  = (__bf16*)take(wDB);
  __bf16* pWk  = (__bf16*)take(wDB);
  __bf16* pWv  = (__bf16*)take(wDB);
  __bf16* pWo  = (__bf16*)take(wDB);
  __bf16* pW1  = (__bf16*)take(wFB);
  __bf16* pW2  = (__bf16*)take(wFB);
  __bf16* qb   = (__bf16*)take(actB);
  __bf16* kb   = (__bf16*)take(actB);
  __bf16* vb   = (__bf16*)take(actB);
  __bf16* vtb  = (__bf16*)take(actB);
  __bf16* ctxb = (__bf16*)take(actB);
  float*  x1f  = (float*)take((size_t)ROWS_ * D_ * sizeof(float));
  __bf16* xn2  = (__bf16*)take(actB);
  __bf16* hb   = (__bf16*)take(ffB);

  // 1) LN1 -> bf16
  k_layernorm<<<ROWS_ / 8, 256, 0, stream>>>(q, ln1s, ln1b, xn);

  // 2) pack weights to bf16 WMMA fragment layout
  k_pack<<<(D_ * D_) / 256, 256, 0, stream>>>(Wq, pWq, D_, D_ / 32, D_ * D_);
  k_pack<<<(D_ * D_) / 256, 256, 0, stream>>>(Wk, pWk, D_, D_ / 32, D_ * D_);
  k_pack<<<(D_ * D_) / 256, 256, 0, stream>>>(Wv, pWv, D_, D_ / 32, D_ * D_);
  k_pack<<<(D_ * D_) / 256, 256, 0, stream>>>(Wo, pWo, D_, D_ / 32, D_ * D_);
  k_pack<<<(D_ * DFF_) / 256, 256, 0, stream>>>(W1, pW1, DFF_, D_ / 32, D_ * DFF_);
  k_pack<<<(D_ * DFF_) / 256, 256, 0, stream>>>(W2, pW2, D_, DFF_ / 32, D_ * DFF_);

  // 3) QKV projections (no bias)
  dim3 gD(ROWS_ / 256, D_ / 64);
  k_gemm<<<gD, 256, 0, stream>>>(xn, pWq, D_, D_, nullptr, nullptr, 0, qb, nullptr);
  k_gemm<<<gD, 256, 0, stream>>>(xn, pWk, D_, D_, nullptr, nullptr, 0, kb, nullptr);
  k_gemm<<<gD, 256, 0, stream>>>(xn, pWv, D_, D_, nullptr, nullptr, 0, vb, nullptr);

  // 4) V transpose for contiguous PV fragments
  k_transpose_v<<<(B_ * H_ * HD_ * S_) / 256, 256, 0, stream>>>(vb, vtb);

  // 5) flash attention
  k_attn<<<(B_ * H_ * (S_ / 16)) / 8, 256, 0, stream>>>(qb, kb, vtb, qmask, ctxb);

  // 6) output projection + residual (x1 = attn_out + q), fp32
  k_gemm<<<gD, 256, 0, stream>>>(ctxb, pWo, D_, D_, bo, q, 0, nullptr, x1f);

  // 7) LN2 -> bf16
  k_layernorm<<<ROWS_ / 8, 256, 0, stream>>>(x1f, ln2s, ln2b, xn2);

  // 8) FFN up + exact GELU -> bf16
  dim3 gF(ROWS_ / 256, DFF_ / 64);
  k_gemm<<<gF, 256, 0, stream>>>(xn2, pW1, D_, DFF_, b1, nullptr, 1, hb, nullptr);

  // 9) FFN down + residual -> fp32 output
  k_gemm<<<gD, 256, 0, stream>>>(hb, pW2, DFF_, D_, b2, x1f, 0, nullptr,
                                 (float*)d_out);
}
